// MultiHeadAttention_84817014161545
// MI455X (gfx1250) — compile-verified, matrix operand order checked by reading, not run on hardware
//
#include <hip/hip_runtime.h>

typedef __attribute__((ext_vector_type(16))) __bf16 v16bf;
typedef __attribute__((ext_vector_type(8)))  __bf16 v8bf;
typedef __attribute__((ext_vector_type(8)))  float  v8f;
typedef __attribute__((ext_vector_type(4)))  float  v4f;

#define EMBED 512
#define HEADS 8
#define HEAD_DIM 64
#define SLEN 4096
#define NB 2
#define MROWS (NB * SLEN)

// ---------------------------------------------------------------------------
// WMMA wrapper
// ---------------------------------------------------------------------------
__device__ inline v8f wmma_bf16(v16bf a, v16bf b, v8f c) {
  return __builtin_amdgcn_wmma_f32_16x16x32_bf16(false, a, false, b, (short)0,
                                                 c, false, false);
}

// ---------------------------------------------------------------------------
// 16-lane butterfly reductions with DPP16 (no LDS traffic).
// ---------------------------------------------------------------------------
__device__ inline float red16_max(float x) {
  x = fmaxf(x, __int_as_float(__builtin_amdgcn_mov_dpp(__float_as_int(x), 0xB1, 0xF, 0xF, true)));
  x = fmaxf(x, __int_as_float(__builtin_amdgcn_mov_dpp(__float_as_int(x), 0x4E, 0xF, 0xF, true)));
  x = fmaxf(x, __int_as_float(__builtin_amdgcn_mov_dpp(__float_as_int(x), 0x141, 0xF, 0xF, true)));
  x = fmaxf(x, __int_as_float(__builtin_amdgcn_mov_dpp(__float_as_int(x), 0x140, 0xF, 0xF, true)));
  return x;
}
__device__ inline float red16_sum(float x) {
  x += __int_as_float(__builtin_amdgcn_mov_dpp(__float_as_int(x), 0xB1, 0xF, 0xF, true));
  x += __int_as_float(__builtin_amdgcn_mov_dpp(__float_as_int(x), 0x4E, 0xF, 0xF, true));
  x += __int_as_float(__builtin_amdgcn_mov_dpp(__float_as_int(x), 0x141, 0xF, 0xF, true));
  x += __int_as_float(__builtin_amdgcn_mov_dpp(__float_as_int(x), 0x140, 0xF, 0xF, true));
  return x;
}

// ---------------------------------------------------------------------------
// Async global->LDS copy (ASYNCcnt) + waits.
// ---------------------------------------------------------------------------
__device__ inline void async_copy_b128(unsigned lds_byte_addr, const void* gaddr) {
  asm volatile("global_load_async_to_lds_b128 %0, %1, off"
               :: "v"(lds_byte_addr), "v"(gaddr) : "memory");
}
__device__ inline void wait_asynccnt0() {
  asm volatile("s_wait_asynccnt 0" ::: "memory");
}

// ---------------------------------------------------------------------------
// 16(row) x 32(k) WMMA operand fragment loader (NT layout), CDNA5 16-bit A/B:
// per lane two contiguous 16B runs at k0 + h*8 and k0 + 16 + h*8, row=lane&15.
// ---------------------------------------------------------------------------
template <typename T>
__device__ inline v16bf load_frag_nt(const T* __restrict__ src, int ld,
                                     int row0, int k0, int lane) {
  const int r = lane & 15;
  const int h = lane >> 4;
  const T* p = src + (size_t)(row0 + r) * ld + k0 + h * 8;
  v16bf f;
  if constexpr (sizeof(T) == 4) {
    const v4f a0 = *(const v4f*)(p);
    const v4f a1 = *(const v4f*)(p + 4);
    const v4f b0 = *(const v4f*)(p + 16);
    const v4f b1 = *(const v4f*)(p + 20);
#pragma unroll
    for (int i = 0; i < 4; ++i) {
      f[i]      = (__bf16)a0[i];
      f[4 + i]  = (__bf16)a1[i];
      f[8 + i]  = (__bf16)b0[i];
      f[12 + i] = (__bf16)b1[i];
    }
  } else {
    const v8bf lo = *(const v8bf*)(p);
    const v8bf hi = *(const v8bf*)(p + 16);
#pragma unroll
    for (int i = 0; i < 8; ++i) {
      f[i]     = lo[i];
      f[8 + i] = hi[i];
    }
  }
  return f;
}

// ---------------------------------------------------------------------------
// NT GEMM: Y[M,N] = yscale * (X[M,K] * W[N,K]^T), W fp32.
// 256 thr = 8 waves; block tile 256(M) x 64(N); wave tile 32 x 64.
// W tile staged once in LDS as bf16 (pitch 520).
// ---------------------------------------------------------------------------
template <typename TA, typename TO>
__global__ __launch_bounds__(256) void gemm_nt(const TA* __restrict__ X,
                                               const float* __restrict__ W,
                                               TO* __restrict__ Y,
                                               int M, int N, int K,
                                               float yscale) {
  constexpr int WP = 520;
  __shared__ __bf16 sW[64 * WP];

  const int lane = threadIdx.x & 31;
  const int wave = threadIdx.x >> 5;
  const int m0 = blockIdx.x * 256 + wave * 32;
  const int n0 = blockIdx.y * 64;

  {
    const int row = threadIdx.x >> 2;
    const int q   = threadIdx.x & 3;
    const float* wr = W + (size_t)(n0 + row) * K;
    for (int k0 = q * 8; k0 < K; k0 += 32) {
      const v4f lo = *(const v4f*)(wr + k0);
      const v4f hi = *(const v4f*)(wr + k0 + 4);
      v8bf pk;
#pragma unroll
      for (int j = 0; j < 4; ++j) { pk[j] = (__bf16)lo[j]; pk[4 + j] = (__bf16)hi[j]; }
      *(v8bf*)&sW[row * WP + k0] = pk;
    }
  }
  __syncthreads();

  v8f acc[2][4] = {};
  for (int k = 0; k < K; k += 32) {
    __builtin_prefetch(X + (size_t)(m0 + (lane & 15)) * K + k + 128, 0, 0);
    const v16bf a0 = load_frag_nt(X, K, m0, k, lane);
    const v16bf a1 = load_frag_nt(X, K, m0 + 16, k, lane);
#pragma unroll
    for (int nt = 0; nt < 4; ++nt) {
      const v16bf b = load_frag_nt((const __bf16*)sW, WP, nt * 16, k, lane);
      acc[0][nt] = wmma_bf16(a0, b, acc[0][nt]);
      acc[1][nt] = wmma_bf16(a1, b, acc[1][nt]);
    }
  }

  const int r15 = lane & 15;
  const int hh  = lane >> 4;
#pragma unroll
  for (int mt = 0; mt < 2; ++mt)
#pragma unroll
    for (int nt = 0; nt < 4; ++nt)
#pragma unroll
      for (int r = 0; r < 8; ++r) {
        const size_t row = (size_t)(m0 + mt * 16 + r + 8 * hh);
        const int    col = n0 + nt * 16 + r15;
        const float  v   = acc[mt][nt][r] * yscale;
        if constexpr (sizeof(TO) == 2) Y[row * N + col] = (__bf16)v;
        else                           Y[row * N + col] = v;
      }
}

// ---------------------------------------------------------------------------
// Flash attention. Grid (SLEN/256, NB*HEADS); 256 thr = 8 waves; wave owns
// 32 query rows (2 q-tiles -> every LDS B-fragment feeds 2 WMMAs).
// Softmax scale is pre-folded into Q by the projection GEMM.
//   Q tile (256x64)  -> LDS once via GLOBAL_LOAD_ASYNC_TO_LDS_B128 (ASYNCcnt);
//                       region is then reused for the per-wave P tiles.
//   K chunk (64x64)  -> LDS via TENSOR_LOAD_TO_LDS (TDM, TENSORcnt), with D#
//                       pad fields producing the 144B LDS pitch; async fallback.
//   V chunk          -> LDS transposed [dim][key] via b128 load + ds_store_b16.
// ---------------------------------------------------------------------------
__global__ __launch_bounds__(256) void flash_attn(const __bf16* __restrict__ Q,
                                                  const __bf16* __restrict__ Kb,
                                                  const __bf16* __restrict__ Vb,
                                                  __bf16* __restrict__ O) {
  constexpr int KP = 72;  // bf16 pitch: 144B rows, 16B-aligned
  __shared__ __align__(16) char smem[256 * KP * 2 + 64 * KP * 2 * 2];
  __bf16* sQP = (__bf16*)smem;                       // Q tile, reused as P
  __bf16* sK  = (__bf16*)(smem + 256 * KP * 2);      // K chunk [key][dim]
  __bf16* sVT = (__bf16*)(smem + 256 * KP * 2 + 64 * KP * 2);  // V^T [dim][key]

  const int lane = threadIdx.x & 31;
  const int wave = threadIdx.x >> 5;
  const int n = blockIdx.y >> 3;
  const int h = blockIdx.y & 7;
  const int qblk = blockIdx.x * 256;

  const __bf16* qp = Q  + (size_t)n * SLEN * EMBED + h * HEAD_DIM;
  const __bf16* kp = Kb + (size_t)n * SLEN * EMBED + h * HEAD_DIM;
  const __bf16* vp = Vb + (size_t)n * SLEN * EMBED + h * HEAD_DIM;

  // ---- stage Q tile (256 rows x 64 dims) into LDS via async copies ----
  {
    const int row = threadIdx.x;
    const __bf16* gq = qp + (size_t)(qblk + row) * EMBED;
    const unsigned lq = (unsigned)(uintptr_t)&sQP[row * KP];
#pragma unroll
    for (int i = 0; i < 8; ++i) async_copy_b128(lq + i * 16, gq + i * 8);
    wait_asynccnt0();
  }
  __syncthreads();

  v16bf fq[2][2];
#pragma unroll
  for (int qt = 0; qt < 2; ++qt) {
    fq[qt][0] = load_frag_nt((const __bf16*)sQP, KP, wave * 32 + qt * 16, 0, lane);
    fq[qt][1] = load_frag_nt((const __bf16*)sQP, KP, wave * 32 + qt * 16, 32, lane);
  }

  float m_r[2][8], l_r[2][8];
  v8f accO[2][4] = {};
#pragma unroll
  for (int qt = 0; qt < 2; ++qt)
#pragma unroll
    for (int r = 0; r < 8; ++r) { m_r[qt][r] = -__builtin_inff(); l_r[qt][r] = 0.0f; }

  const float L2E = 1.4426950408889634f;
  __bf16* myP = sQP + wave * 32 * KP;  // aliases the (dead) Q staging rows

  for (int kc = 0; kc < SLEN; kc += 64) {
    __syncthreads();  // previous chunk's compute done before restaging
    // --- K chunk: Tensor Data Mover (one issue from wave 0) ---
#if __has_builtin(__builtin_amdgcn_tensor_load_to_lds)
    if (wave == 0) {
      typedef unsigned uv4 __attribute__((ext_vector_type(4)));
      typedef int      iv4 __attribute__((ext_vector_type(4)));
      typedef int      iv8 __attribute__((ext_vector_type(8)));
      const unsigned long long ga =
          (unsigned long long)(uintptr_t)(kp + (size_t)kc * EMBED);
      const uv4 g0 = {
          1u,                                     // count=1 (user descriptor)
          (unsigned)(uintptr_t)sK,                // lds_addr
          (unsigned)(ga & 0xFFFFFFFFu),           // global_addr lo
          (unsigned)((ga >> 32) & 0x01FFFFFFu) | (2u << 30)};  // hi | type=2
      const iv8 g1 = {
          (int)((1u << 16) | (1u << 20) | (4u << 22) | (3u << 25)),
          //     data_size=2B  pad_en     every 32dw    pad 4dw (128B->+16B = KP)
          (int)(64u << 16),   // tensor_dim0 = 64
          (int)(64u << 16),   // tensor_dim1 = 64
          (int)(64u << 16),   // tile_dim0 = 64
          (int)64u,           // tile_dim1 = 64, tile_dim2 = 0
          (int)512u,          // tensor_dim0_stride = EMBED
          0, 0};
      const iv4 gz = {0, 0, 0, 0};
#if __clang_major__ >= 23
      const iv8 gz8 = {0, 0, 0, 0, 0, 0, 0, 0};
      __builtin_amdgcn_tensor_load_to_lds(g0, g1, gz, gz, gz8, 0);
#else
      __builtin_amdgcn_tensor_load_to_lds(g0, g1, gz, gz, 0);
#endif
      __builtin_amdgcn_s_wait_tensorcnt(0);
    }
#else
    {
      const int key = threadIdx.x >> 2;
      const int d0  = (threadIdx.x & 3) * 16;
      const __bf16* gk = kp + (size_t)(kc + key) * EMBED + d0;
      const unsigned lk = (unsigned)(uintptr_t)&sK[key * KP + d0];
      async_copy_b128(lk, gk);
      async_copy_b128(lk + 16, gk + 8);
    }
#endif
    // --- V chunk: load + transpose-scatter ---
    {
      const int key = threadIdx.x >> 2;        // 0..63
      const int d0  = (threadIdx.x & 3) * 16;  // 0,16,32,48
      const __bf16* gv = vp + (size_t)(kc + key) * EMBED + d0;
      const v8bf v0 = *(const v8bf*)(gv);
      const v8bf v1 = *(const v8bf*)(gv + 8);
#pragma unroll
      for (int i = 0; i < 8; ++i) {
        sVT[(d0 + i) * KP + key]     = v0[i];
        sVT[(d0 + 8 + i) * KP + key] = v1[i];
      }
#if !__has_builtin(__builtin_amdgcn_tensor_load_to_lds)
      wait_asynccnt0();
#endif
    }
    __syncthreads();

    // ---- energy: 4 key-subtiles x 2 q-tiles, B fragments reused ----
    v8f e[2][4];
#pragma unroll
    for (int t = 0; t < 4; ++t) {
      const v16bf bk0 = load_frag_nt((const __bf16*)sK, KP, t * 16, 0, lane);
      const v16bf bk1 = load_frag_nt((const __bf16*)sK, KP, t * 16, 32, lane);
#pragma unroll
      for (int qt = 0; qt < 2; ++qt) {
        v8f c = {};
        c = wmma_bf16(fq[qt][0], bk0, c);
        c = wmma_bf16(fq[qt][1], bk1, c);
        e[qt][t] = c;
      }
    }

    // ---- online softmax (scale already folded into Q) ----
#pragma unroll
    for (int qt = 0; qt < 2; ++qt) {
      __bf16* pP = myP + qt * 16 * KP;
#pragma unroll
      for (int r = 0; r < 8; ++r) {
        float mx = fmaxf(fmaxf(e[qt][0][r], e[qt][1][r]),
                         fmaxf(e[qt][2][r], e[qt][3][r]));
        mx = red16_max(mx);
        const float mnew = fmaxf(m_r[qt][r], mx);
        const int row = r + 8 * (lane >> 4);
        float ps = 0.0f;
#pragma unroll
        for (int t = 0; t < 4; ++t) {
          const float p = exp2f((e[qt][t][r] - mnew) * L2E);
          ps += p;
          pP[row * KP + t * 16 + (lane & 15)] = (__bf16)p;  // C->A via LDS
        }
        ps = red16_sum(ps);
        const float alpha = exp2f((m_r[qt][r] - mnew) * L2E);
        l_r[qt][r] = l_r[qt][r] * alpha + ps;
        m_r[qt][r] = mnew;
#pragma unroll
        for (int nt = 0; nt < 4; ++nt) accO[qt][nt][r] *= alpha;
      }
    }
    __builtin_amdgcn_wave_barrier();  // DS in-order per wave; block reordering

    // ---- P x V: V fragments reused across q-tiles ----
    v16bf fp[2][2];
#pragma unroll
    for (int qt = 0; qt < 2; ++qt) {
      fp[qt][0] = load_frag_nt((const __bf16*)myP, KP, qt * 16, 0, lane);
      fp[qt][1] = load_frag_nt((const __bf16*)myP, KP, qt * 16, 32, lane);
    }
    __builtin_amdgcn_wave_barrier();
#pragma unroll
    for (int nt = 0; nt < 4; ++nt) {
      const v16bf fv0 = load_frag_nt((const __bf16*)sVT, KP, nt * 16, 0, lane);
      const v16bf fv1 = load_frag_nt((const __bf16*)sVT, KP, nt * 16, 32, lane);
#pragma unroll
      for (int qt = 0; qt < 2; ++qt) {
        accO[qt][nt] = wmma_bf16(fp[qt][0], fv0, accO[qt][nt]);
        accO[qt][nt] = wmma_bf16(fp[qt][1], fv1, accO[qt][nt]);
      }
    }
  }

  // ---- epilogue ----
  const int r15 = lane & 15;
  const int hh  = lane >> 4;
#pragma unroll
  for (int qt = 0; qt < 2; ++qt)
#pragma unroll
    for (int r = 0; r < 8; ++r) {
      const float inv = 1.0f / l_r[qt][r];
      const int qrow = qblk + wave * 32 + qt * 16 + r + 8 * hh;
      __bf16* op = O + ((size_t)n * SLEN + qrow) * EMBED + h * HEAD_DIM;
#pragma unroll
      for (int nt = 0; nt < 4; ++nt)
        op[nt * 16 + r15] = (__bf16)(accO[qt][nt][r] * inv);
    }
}

// ---------------------------------------------------------------------------
extern "C" void kernel_launch(void* const* d_in, const int* in_sizes, int n_in,
                              void* d_out, int out_size, void* d_ws,
                              size_t ws_size, hipStream_t stream) {
  const float* in_v = (const float*)d_in[0];
  const float* in_q = (const float*)d_in[1];
  const float* in_k = (const float*)d_in[2];
  const float* W_Q  = (const float*)d_in[3];
  const float* W_K  = (const float*)d_in[4];
  const float* W_V  = (const float*)d_in[5];
  const float* W_fc = (const float*)d_in[6];
  float* out = (float*)d_out;

  __bf16* qb = (__bf16*)d_ws;
  __bf16* kb = qb + (size_t)MROWS * EMBED;
  __bf16* vb = kb + (size_t)MROWS * EMBED;
  __bf16* ob = vb + (size_t)MROWS * EMBED;

  const float scale = 0.044194173824159216f;  // 1/sqrt(EMBED), folded into Q
  dim3 gg(MROWS / 256, EMBED / 64), bb(256);
  gemm_nt<float, __bf16><<<gg, bb, 0, stream>>>(in_q, W_Q, qb, MROWS, EMBED, EMBED, scale);
  gemm_nt<float, __bf16><<<gg, bb, 0, stream>>>(in_k, W_K, kb, MROWS, EMBED, EMBED, 1.0f);
  gemm_nt<float, __bf16><<<gg, bb, 0, stream>>>(in_v, W_V, vb, MROWS, EMBED, EMBED, 1.0f);

  flash_attn<<<dim3(SLEN / 256, NB * HEADS), 256, 0, stream>>>(qb, kb, vb, ob);

  gemm_nt<__bf16, float><<<gg, bb, 0, stream>>>(ob, W_fc, out, MROWS, EMBED, EMBED, 1.0f);
}
